// BipartiteGCN_15178414424447
// MI455X (gfx1250) — matrix operand order repaired and checked by reading, not compiled, not run on hardware
//
#include <hip/hip_runtime.h>
#include <hip/hip_bf16.h>

// ---------------------------------------------------------------------------
// BipartiteGCN on MI455X (gfx1250, wave32, WMMA).
// All matmuls lowered to V_WMMA_F32_16X16X32_F16 (f16 in, f32 accumulate):
//   - edge message MLP  [1M x 64] @ [64 x 64]   (dominant FLOPs, L2-resident)
//   - node MLPs         [N x {32,64,128}pad] @ [K x 64]
// Weight tiles staged global->LDS with GLOBAL_LOAD_ASYNC_TO_LDS_B128
// (ASYNCcnt). Scatter epilogues are two-phase and fully branchless: tail
// edges are redirected to a spare sink row via a 32-bit index select, and
// per-row base addresses are hoisted so inner atomics use immediate offsets.
// ---------------------------------------------------------------------------

#define EMB     64
#define NCONS   50000
#define NVARS   100000
#define NEDGE   1000000
#define CONS_NF 5
#define VAR_NF  43
#define SLOPE   0.01f
#define LNEPS   1e-5f

typedef __attribute__((ext_vector_type(16))) _Float16 v16h;
typedef __attribute__((ext_vector_type(8)))  _Float16 v8h;
typedef __attribute__((ext_vector_type(8)))  float    v8f;

__device__ __forceinline__ v8f wmma16(v16h a, v16h b, v8f c) {
    // (neg_a, A, neg_b, B, c_mod, C, reuse_a, reuse_b)
    return __builtin_amdgcn_wmma_f32_16x16x32_f16(false, a, false, b, (short)0, c,
                                                  false, false);
}
__device__ __forceinline__ float leakyf(float x) { return x >= 0.f ? x : SLOPE * x; }

// async global->LDS copy, 16 bytes per lane (VDST = LDS byte offset VGPR)
__device__ __forceinline__ void async_copy16(unsigned lds_off, const void* gptr) {
    asm volatile("global_load_async_to_lds_b128 %0, %1, off"
                 :: "v"(lds_off), "v"(gptr) : "memory");
}
__device__ __forceinline__ void wait_async0() {
    asm volatile("s_wait_asynccnt 0x0" ::: "memory");
}

// ---------------------------------------------------------------------------
// Weight prep: W[Kin][64] f32 (row-major, in x out)  ->  WT[64][Kpad] f16
// ---------------------------------------------------------------------------
__global__ void k_prep_wt(const float* __restrict__ W, int Kin, int Kpad,
                          _Float16* __restrict__ WT) {
    int i = blockIdx.x * blockDim.x + threadIdx.x;
    int tot = 64 * Kpad;
    if (i >= tot) return;
    int n = i / Kpad, k = i % Kpad;
    float v = (k < Kin) ? W[k * 64 + n] : 0.f;
    WT[i] = (_Float16)v;
}

__global__ void k_zero(float* __restrict__ p, int n) {
    int i = blockIdx.x * blockDim.x + threadIdx.x;
    if (i < n) p[i] = 0.f;
}

__global__ void k_count(const int* __restrict__ dst, int ne, float* __restrict__ cnt) {
    int i = blockIdx.x * blockDim.x + threadIdx.x;
    if (i < ne) atomicAdd(&cnt[dst[i]], 1.f);
}

// ---------------------------------------------------------------------------
// Generic tall-skinny GEMM:  Y[n x 64] = act( (pre_ln(X))[n x Kin] @ W + bias )
// KP = padded K (32/64/128). 256 threads = 8 waves, 16 rows/wave, 128 rows/blk.
// ---------------------------------------------------------------------------
template <int KP>
__global__ __launch_bounds__(256) void k_rows_gemm(
    const float* __restrict__ X, int nrows, int Kin,
    const float* __restrict__ lng, const float* __restrict__ lnb,
    const _Float16* __restrict__ WT, const float* __restrict__ bias,
    int postLeaky, float* __restrict__ Y, float* __restrict__ sink) {
    __shared__ _Float16 sW[64 * KP];
    __shared__ _Float16 sA[128 * KP];

    const int tid  = threadIdx.x;
    const int lane = tid & 31;
    const int wv   = tid >> 5;
    const int hf   = lane >> 4;   // which half of the wave (0/1)
    const int l16  = lane & 15;

    // stage weights (f16) into LDS via async DMA (ASYNCcnt)
    {
        const char* g = (const char*)WT;
        const unsigned l0 = (unsigned)(uintptr_t)(&sW[0]);
        for (int b = tid * 16; b < 64 * KP * 2; b += 256 * 16)
            async_copy16(l0 + (unsigned)b, g + b);
        wait_async0();
    }

    const int rowBlock = blockIdx.x * 128;
    const int row = rowBlock + wv * 16 + l16;
    _Float16* aRow = &sA[(wv * 16 + l16) * KP];
    const int cpl = KP / 2;          // columns handled per lane
    const int c0  = hf * cpl;

    if (lng) {  // fused LayerNorm over Kin, then f16 convert
        float s1 = 0.f, s2 = 0.f;
        for (int i = 0; i < cpl; i++) {
            int c = c0 + i;
            float x = (row < nrows && c < Kin) ? X[(size_t)row * Kin + c] : 0.f;
            s1 += x; s2 += x * x;
        }
        s1 += __shfl_xor(s1, 16, 32);
        s2 += __shfl_xor(s2, 16, 32);
        float mn   = s1 / (float)Kin;
        float var  = s2 / (float)Kin - mn * mn;
        float rstd = rsqrtf(var + LNEPS);
        for (int i = 0; i < cpl; i++) {
            int c = c0 + i;
            float x = (row < nrows && c < Kin) ? X[(size_t)row * Kin + c] : 0.f;
            float t = (c < Kin) ? ((x - mn) * rstd * lng[c] + lnb[c]) : 0.f;
            aRow[c] = (_Float16)t;
        }
    } else {
        for (int i = 0; i < cpl; i++) {
            int c = c0 + i;
            float x = (row < nrows && c < Kin) ? X[(size_t)row * Kin + c] : 0.f;
            aRow[c] = (_Float16)x;
        }
    }
    __syncthreads();

    // A fragments (16x32 f16 per chunk): lane(l16,hf) = row l16,
    // K = chunk*32 + hf*8 + {0..7} and + 16 + {0..7}
    v16h afr[KP / 32];
#pragma unroll
    for (int ch = 0; ch < KP / 32; ch++) {
        const _Float16* p = aRow + ch * 32 + hf * 8;
        v8h lo = *(const v8h*)p;
        v8h hi = *(const v8h*)(p + 16);
        v16h a;
#pragma unroll
        for (int i = 0; i < 8; i++) { a[i] = lo[i]; a[i + 8] = hi[i]; }
        afr[ch] = a;
    }

    // phase 1: all four 16x16 accumulator tiles
    v8f acc4[4];
#pragma unroll
    for (int nt = 0; nt < 4; nt++) {
        v8f acc = {};
        const int n = nt * 16 + l16;
#pragma unroll
        for (int ch = 0; ch < KP / 32; ch++) {
            // B fragment: lane holds col n, K = ch*32 + hf*8 + {0..7} and + 16 + {0..7}
            const _Float16* q = &sW[n * KP + ch * 32 + hf * 8];
            v16h b = __builtin_shufflevector(*(const v8h*)q, *(const v8h*)(q + 16), 0, 1, 2, 3, 4, 5, 6, 7, 8, 9, 10, 11, 12, 13, 14, 15);
            acc = wmma16(afr[ch], b, acc);
        }
        acc4[nt] = acc;
    }

    float bv4[4];
#pragma unroll
    for (int nt = 0; nt < 4; nt++) bv4[nt] = bias[nt * 16 + l16];

    // phase 2: stores with hoisted per-row base (branchless via sink line)
#pragma unroll
    for (int r = 0; r < 8; r++) {
        int orow = rowBlock + wv * 16 + hf * 8 + r;   // C/D layout: M = r + 8*hf
        float* ybase = (orow < nrows) ? (Y + (size_t)orow * 64 + l16) : (sink + l16);
#pragma unroll
        for (int nt = 0; nt < 4; nt++) {
            float y = acc4[nt][r] + bv4[nt];
            if (postLeaky) y = leakyf(y);
            ybase[nt * 16] = y;
        }
    }
}

// ---------------------------------------------------------------------------
// Fused edge-message kernel:
//   msg = LeakyReLU(LN(lf[src] + rf[dst])) @ Wf + bf ;  S[dst] += msg (atomic)
// 16 edges / wave, 8 waves / block. S has (nr+1) rows; row `nr` is the sink
// row for the tail-guard so the scatter is fully branchless.
// ---------------------------------------------------------------------------
__global__ __launch_bounds__(256) void k_edge_msg(
    const float* __restrict__ lf, const float* __restrict__ rf,
    const int* __restrict__ src, const int* __restrict__ dst, int ne,
    const float* __restrict__ lng, const float* __restrict__ lnb,
    const _Float16* __restrict__ WfT, const float* __restrict__ bias,
    float* __restrict__ S, int sinkRow) {
    __shared__ _Float16 sW[64 * 64];
    __shared__ _Float16 sA[128 * 64];

    const int tid  = threadIdx.x;
    const int lane = tid & 31;
    const int wv   = tid >> 5;
    const int hf   = lane >> 4;
    const int l16  = lane & 15;

    // async-stage Wf (8 KB) into LDS
    {
        const char* g = (const char*)WfT;
        const unsigned l0 = (unsigned)(uintptr_t)(&sW[0]);
        for (int b = tid * 16; b < 64 * 64 * 2; b += 256 * 16)
            async_copy16(l0 + (unsigned)b, g + b);
        wait_async0();
    }

    const int e0 = blockIdx.x * 128 + wv * 16;   // this wave's 16-edge tile
    const int e  = e0 + l16;
    const bool ev = (e < ne);
    int si = ev ? src[e] : 0;
    int di = ev ? dst[e] : 0;

    const int c0 = hf * 32;
    float xv[32];
    const float4* pl = (const float4*)(lf + (size_t)si * 64 + c0);
    const float4* pr = (const float4*)(rf + (size_t)di * 64 + c0);
    float s1 = 0.f, s2 = 0.f;
#pragma unroll
    for (int i = 0; i < 8; i++) {
        float4 a = pl[i], b = pr[i];
        float v0 = a.x + b.x, v1 = a.y + b.y, v2 = a.z + b.z, v3 = a.w + b.w;
        xv[4 * i] = v0; xv[4 * i + 1] = v1; xv[4 * i + 2] = v2; xv[4 * i + 3] = v3;
        s1 += v0 + v1 + v2 + v3;
        s2 += v0 * v0 + v1 * v1 + v2 * v2 + v3 * v3;
    }
    s1 += __shfl_xor(s1, 16, 32);
    s2 += __shfl_xor(s2, 16, 32);
    float mn   = s1 * (1.f / 64.f);
    float var  = s2 * (1.f / 64.f) - mn * mn;
    float rstd = rsqrtf(var + LNEPS);

    _Float16* aRow = &sA[(wv * 16 + l16) * 64];
#pragma unroll
    for (int i = 0; i < 32; i++) {
        int c = c0 + i;
        float t = (xv[i] - mn) * rstd * lng[c] + lnb[c];
        aRow[c] = (_Float16)leakyf(t);
    }
    __syncthreads();

    v16h afr[2];
#pragma unroll
    for (int ch = 0; ch < 2; ch++) {
        const _Float16* p = aRow + ch * 32 + hf * 8;
        v8h lo = *(const v8h*)p;
        v8h hi = *(const v8h*)(p + 16);
        v16h a;
#pragma unroll
        for (int i = 0; i < 8; i++) { a[i] = lo[i]; a[i + 8] = hi[i]; }
        afr[ch] = a;
    }

    // destination rows for this tile: broadcast from staging registers
    // (lane l16 holds dst[e0+l16]); tail edges -> sink row (int cndmask)
    int dj8[8];
#pragma unroll
    for (int r = 0; r < 8; r++) {
        int mrow = hf * 8 + r;
        int dj = __shfl(di, mrow, 32);
        dj8[r] = ((e0 + mrow) < ne) ? dj : sinkRow;
    }

    // phase 1: all four 16x16 accumulator tiles
    v8f acc4[4];
#pragma unroll
    for (int nt = 0; nt < 4; nt++) {
        v8f acc = {};
#pragma unroll
        for (int ch = 0; ch < 2; ch++) {
            const _Float16* q = &sW[(nt * 16 + l16) * 64 + ch * 32 + hf * 8];
            v16h b = __builtin_shufflevector(*(const v8h*)q, *(const v8h*)(q + 16), 0, 1, 2, 3, 4, 5, 6, 7, 8, 9, 10, 11, 12, 13, 14, 15);
            acc = wmma16(afr[ch], b, acc);
        }
        acc4[nt] = acc;
    }

    float bv4[4];
#pragma unroll
    for (int nt = 0; nt < 4; nt++) bv4[nt] = bias[nt * 16 + l16];

    // phase 2: scatter; row base computed once per r, inner offsets immediate
#pragma unroll
    for (int r = 0; r < 8; r++) {
        float* base = S + (size_t)dj8[r] * 64 + l16;
#pragma unroll
        for (int nt = 0; nt < 4; nt++) {
            atomicAdd(base + nt * 16, acc4[nt][r] + bv4[nt]);
        }
    }
}

// ---------------------------------------------------------------------------
// Per-node: agg = S/max(cnt,1); H = [LN(agg) , right]   (one 64-thread block)
// ---------------------------------------------------------------------------
__global__ void k_agg_concat(const float* __restrict__ S, const float* __restrict__ cnt,
                             const float* __restrict__ right,
                             const float* __restrict__ g, const float* __restrict__ b,
                             float* __restrict__ H, int nr) {
    __shared__ float sh[64];
    int node = blockIdx.x;
    if (node >= nr) return;
    int t = threadIdx.x;
    float a = S[(size_t)node * 64 + t] / fmaxf(cnt[node], 1.f);
    sh[t] = a; __syncthreads();
    for (int o = 32; o > 0; o >>= 1) { if (t < o) sh[t] += sh[t + o]; __syncthreads(); }
    float mn = sh[0] * (1.f / 64.f); __syncthreads();
    float d = a - mn;
    sh[t] = d * d; __syncthreads();
    for (int o = 32; o > 0; o >>= 1) { if (t < o) sh[t] += sh[t + o]; __syncthreads(); }
    float var = sh[0] * (1.f / 64.f);
    float p = d * rsqrtf(var + LNEPS) * g[t] + b[t];
    H[(size_t)node * 128 + t] = p;
    H[(size_t)node * 128 + 64 + t] = right[(size_t)node * 64 + t];
}

// final head: out[node] = T[node] . h2W + h2b
__global__ void k_head(const float* __restrict__ T, const float* __restrict__ w,
                       const float* __restrict__ b, float* __restrict__ out, int n) {
    __shared__ float sh[64];
    int node = blockIdx.x;
    if (node >= n) return;
    int t = threadIdx.x;
    sh[t] = T[(size_t)node * 64 + t] * w[t];
    __syncthreads();
    for (int o = 32; o > 0; o >>= 1) { if (t < o) sh[t] += sh[t + o]; __syncthreads(); }
    if (t == 0) out[node] = sh[0] + b[0];
}

// ---------------------------------------------------------------------------

struct EmbP  { const float *W1, *W2, *b1, *b2, *ln_b, *ln_g; };
struct ConvP { const float *Wf, *Wl, *Wo1, *Wo2, *Wr, *bf, *bl, *bo1, *bo2,
                           *fln_b, *fln_g, *pln_b, *pln_g; };

extern "C" void kernel_launch(void* const* d_in, const int* in_sizes, int n_in,
                              void* d_out, int out_size, void* d_ws, size_t ws_size,
                              hipStream_t stream) {
    (void)in_sizes; (void)n_in; (void)out_size; (void)ws_size;

    // Input order: top-level setup_inputs() dict in insertion order; the
    // 'params' pytree expanded at its slot in JAX sorted-key order.
    const float* cons_feats = (const float*)d_in[0];
    const float* var_feats  = (const float*)d_in[1];
    // d_in[2] = edge_vals (unused by the reference math)
    int ix = 3;
    auto nxt = [&]() { return (const float*)d_in[ix++]; };
    auto getEmb = [&]() {
        EmbP p; p.W1 = nxt(); p.W2 = nxt(); p.b1 = nxt(); p.b2 = nxt();
        p.ln_b = nxt(); p.ln_g = nxt(); return p;
    };
    auto getConv = [&]() {
        ConvP p; p.Wf = nxt(); p.Wl = nxt(); p.Wo1 = nxt(); p.Wo2 = nxt();
        p.Wr = nxt(); p.bf = nxt(); p.bl = nxt(); p.bo1 = nxt(); p.bo2 = nxt();
        p.fln_b = nxt(); p.fln_g = nxt(); p.pln_b = nxt(); p.pln_g = nxt(); return p;
    };
    // sorted keys: cons_emb, ctov, h1W, h1b, h2W, h2b, var_emb, vtoc
    EmbP  ce   = getEmb();
    ConvP ctov = getConv();
    const float* h1W = nxt();
    const float* h1b = nxt();
    const float* h2W = nxt();
    const float* h2b = nxt();
    EmbP  ve   = getEmb();
    ConvP vtoc = getConv();
    const int* edge_cons = (const int*)d_in[ix++];
    const int* edge_vars = (const int*)d_in[ix++];

    // --- carve workspace (256B aligned) ---
    char* ws = (char*)d_ws;
    size_t off = 0;
    auto carve = [&](size_t bytes) -> void* {
        void* p = ws + off;
        off = (off + bytes + 255) & ~(size_t)255;
        return p;
    };
    _Float16* cW1T   = (_Float16*)carve(64 * 32 * 2);
    _Float16* cW2T   = (_Float16*)carve(64 * 64 * 2);
    _Float16* vW1T   = (_Float16*)carve(64 * 64 * 2);
    _Float16* vW2T   = (_Float16*)carve(64 * 64 * 2);
    _Float16* vtWlT  = (_Float16*)carve(64 * 64 * 2);
    _Float16* vtWrT  = (_Float16*)carve(64 * 64 * 2);
    _Float16* vtWfT  = (_Float16*)carve(64 * 64 * 2);
    _Float16* vtWo1T = (_Float16*)carve(64 * 128 * 2);
    _Float16* vtWo2T = (_Float16*)carve(64 * 64 * 2);
    _Float16* cvWlT  = (_Float16*)carve(64 * 64 * 2);
    _Float16* cvWrT  = (_Float16*)carve(64 * 64 * 2);
    _Float16* cvWfT  = (_Float16*)carve(64 * 64 * 2);
    _Float16* cvWo1T = (_Float16*)carve(64 * 128 * 2);
    _Float16* cvWo2T = (_Float16*)carve(64 * 64 * 2);
    _Float16* h1WT   = (_Float16*)carve(64 * 64 * 2);
    float* zeros64 = (float*)carve(64 * 4);
    float* sink    = (float*)carve(128 * 4);
    float* cnt_c   = (float*)carve((size_t)NCONS * 4);
    float* cnt_v   = (float*)carve((size_t)NVARS * 4);
    float* varh0   = (float*)carve((size_t)NVARS * 64 * 4);
    float* varh1   = (float*)carve((size_t)NVARS * 64 * 4);
    float* cons0   = (float*)carve((size_t)NCONS * 64 * 4);
    float* cons1   = (float*)carve((size_t)NCONS * 64 * 4);
    float* lfB     = (float*)carve((size_t)NVARS * 64 * 2 * 4);  // lf + rf contiguous
    float* rfB     = lfB + (size_t)NVARS * 64;
    float* SB      = (float*)carve(((size_t)NVARS + 1) * 64 * 4); // +1 sink row
    float* HB      = lfB;  // [n x 128], reuses lf/rf after edge stage
    float* TB      = SB;   // [n x 64],  reuses S after aggregation

    auto prep = [&](const float* W, int Kin, int Kpad, _Float16* WT) {
        int tot = 64 * Kpad;
        k_prep_wt<<<(tot + 255) / 256, 256, 0, stream>>>(W, Kin, Kpad, WT);
    };
    auto zero = [&](float* p, size_t n) {
        k_zero<<<(int)((n + 255) / 256), 256, 0, stream>>>(p, (int)n);
    };

    // weight prep
    prep(ce.W1, CONS_NF, 32, cW1T);   prep(ce.W2, 64, 64, cW2T);
    prep(ve.W1, VAR_NF, 64, vW1T);    prep(ve.W2, 64, 64, vW2T);
    prep(vtoc.Wl, 64, 64, vtWlT);     prep(vtoc.Wr, 64, 64, vtWrT);
    prep(vtoc.Wf, 64, 64, vtWfT);     prep(vtoc.Wo1, 128, 128, vtWo1T);
    prep(vtoc.Wo2, 64, 64, vtWo2T);
    prep(ctov.Wl, 64, 64, cvWlT);     prep(ctov.Wr, 64, 64, cvWrT);
    prep(ctov.Wf, 64, 64, cvWfT);     prep(ctov.Wo1, 128, 128, cvWo1T);
    prep(ctov.Wo2, 64, 64, cvWo2T);
    prep(h1W, 64, 64, h1WT);
    zero(zeros64, 64);

    // degree counts
    zero(cnt_c, NCONS);
    zero(cnt_v, NVARS);
    k_count<<<(NEDGE + 255) / 256, 256, 0, stream>>>(edge_cons, NEDGE, cnt_c);
    k_count<<<(NEDGE + 255) / 256, 256, 0, stream>>>(edge_vars, NEDGE, cnt_v);

    // node embeddings: LN -> Linear -> LeakyReLU -> Linear -> LeakyReLU
    k_rows_gemm<32><<<(NCONS + 127) / 128, 256, 0, stream>>>(
        cons_feats, NCONS, CONS_NF, ce.ln_g, ce.ln_b, cW1T, ce.b1, 1, TB, sink);
    k_rows_gemm<64><<<(NCONS + 127) / 128, 256, 0, stream>>>(
        TB, NCONS, 64, nullptr, nullptr, cW2T, ce.b2, 1, cons0, sink);
    k_rows_gemm<64><<<(NVARS + 127) / 128, 256, 0, stream>>>(
        var_feats, NVARS, VAR_NF, ve.ln_g, ve.ln_b, vW1T, ve.b1, 1, TB, sink);
    k_rows_gemm<64><<<(NVARS + 127) / 128, 256, 0, stream>>>(
        TB, NVARS, 64, nullptr, nullptr, vW2T, ve.b2, 1, varh0, sink);

    auto run_conv = [&](const ConvP& p, const float* left, int nl,
                        const float* right, int nr,
                        const int* esrc, const int* edst, const float* cnt,
                        _Float16* WlT, _Float16* WrT, _Float16* WfT,
                        _Float16* Wo1T, _Float16* Wo2T, float* outBuf) {
        k_rows_gemm<64><<<(nl + 127) / 128, 256, 0, stream>>>(
            left, nl, 64, nullptr, nullptr, WlT, p.bl, 0, lfB, sink);
        k_rows_gemm<64><<<(nr + 127) / 128, 256, 0, stream>>>(
            right, nr, 64, nullptr, nullptr, WrT, zeros64, 0, rfB, sink);
        zero(SB, ((size_t)nr + 1) * 64);
        k_edge_msg<<<(NEDGE + 127) / 128, 256, 0, stream>>>(
            lfB, rfB, esrc, edst, NEDGE, p.fln_g, p.fln_b, WfT, p.bf, SB, nr);
        k_agg_concat<<<nr, 64, 0, stream>>>(SB, cnt, right, p.pln_g, p.pln_b, HB, nr);
        k_rows_gemm<128><<<(nr + 127) / 128, 256, 0, stream>>>(
            HB, nr, 128, nullptr, nullptr, Wo1T, p.bo1, 1, TB, sink);
        k_rows_gemm<64><<<(nr + 127) / 128, 256, 0, stream>>>(
            TB, nr, 64, nullptr, nullptr, Wo2T, p.bo2, 0, outBuf, sink);
    };

    // conv v->c: source=var, target=cons
    run_conv(vtoc, varh0, NVARS, cons0, NCONS, edge_vars, edge_cons, cnt_c,
             vtWlT, vtWrT, vtWfT, vtWo1T, vtWo2T, cons1);
    // conv c->v: source=cons, target=var
    run_conv(ctov, cons1, NCONS, varh0, NVARS, edge_cons, edge_vars, cnt_v,
             cvWlT, cvWrT, cvWfT, cvWo1T, cvWo2T, varh1);

    // head: Linear -> LeakyReLU -> Linear(64->1)
    k_rows_gemm<64><<<(NVARS + 127) / 128, 256, 0, stream>>>(
        varh1, NVARS, 64, nullptr, nullptr, h1WT, h1b, 1, TB, sink);
    k_head<<<NVARS, 64, 0, stream>>>(TB, h2W, h2b, (float*)d_out, NVARS);
}
